// CTI_toC_46789373723362
// MI455X (gfx1250) — matrix operand order repaired and checked by reading, not compiled, not run on hardware
//
#include <hip/hip_runtime.h>

// ---------------- problem constants (fixed by the reference) ----------------
#define DIMC   384
#define HEADS  6
#define HDIM   64          // DIMC / HEADS
#define LEVELS 3
#define POINTS 4
#define HIDC   96
#define BATCH  2
#define NMID   1728        // 12*12*12
#define NTOK   15768       // 73*NMID/8
#define NHIGH  13824       // 8*NMID (level0 tokens)
#define NMIDE  15552       // NHIGH+NMID
#define MROWS  (BATCH*NTOK)  // 31536
#define MPAD   31616         // MROWS padded up to a multiple of 128

typedef __attribute__((ext_vector_type(16))) _Float16 v16h;
typedef __attribute__((ext_vector_type(8)))  float    v8f;

__device__ __forceinline__ float wave_sum32(float v) {
    v += __shfl_xor(v, 16, 32);
    v += __shfl_xor(v, 8, 32);
    v += __shfl_xor(v, 4, 32);
    v += __shfl_xor(v, 2, 32);
    v += __shfl_xor(v, 1, 32);
    return v;
}

// ---------------------------------------------------------------------------
// Weight convert: f32 row-major [K,N]  ->  f16 column-major [N,K]
// ---------------------------------------------------------------------------
__global__ void convert_wt_kernel(const float* __restrict__ w,
                                  _Float16* __restrict__ wt, int K, int N) {
    int t = blockIdx.x * blockDim.x + threadIdx.x;
    if (t >= K * N) return;
    int n = t / K, k = t - n * K;
    wt[(size_t)n * K + k] = (_Float16)w[(size_t)k * N + n];
}

// ---------------------------------------------------------------------------
// Prep: q = query (+feat on mid band); qn=LN(q;qw), fn=LN(q;fw) (shared stats);
// aq=LN(qn;eqw) -> f16, af=LN(fn;efw) -> f16.   One wave per row.
// ---------------------------------------------------------------------------
__global__ __launch_bounds__(256) void prep_kernel(
    const float* __restrict__ query, const float* __restrict__ feat,
    const float* __restrict__ qw, const float* __restrict__ qb,
    const float* __restrict__ fw, const float* __restrict__ fb,
    const float* __restrict__ eqw, const float* __restrict__ eqb,
    const float* __restrict__ efw, const float* __restrict__ efb,
    float* __restrict__ qn_out, _Float16* __restrict__ aq_out,
    _Float16* __restrict__ af_out) {
    int wid = blockIdx.x * (blockDim.x >> 5) + (threadIdx.x >> 5);
    if (wid >= MROWS) return;
    int lane = threadIdx.x & 31;
    int b = wid / NTOK, s = wid - b * NTOK;
    const float* qrow = query + (size_t)wid * DIMC;
    const float* frow = (s >= NHIGH && s < NMIDE)
                            ? feat + ((size_t)b * NMID + (s - NHIGH)) * DIMC
                            : nullptr;
    float x[12];
    float sum = 0.f, sq = 0.f;
#pragma unroll
    for (int j = 0; j < 12; ++j) {
        int c = lane + 32 * j;
        float v = qrow[c];
        if (frow) v += frow[c];
        x[j] = v; sum += v; sq += v * v;
    }
    sum = wave_sum32(sum); sq = wave_sum32(sq);
    float mean = sum * (1.0f / DIMC);
    float var = sq * (1.0f / DIMC) - mean * mean;
    float rs = rsqrtf(var + 1e-6f);

    float qnv[12], fnv[12];
    float s1 = 0.f, s1q = 0.f, s2 = 0.f, s2q = 0.f;
#pragma unroll
    for (int j = 0; j < 12; ++j) {
        int c = lane + 32 * j;
        float nm = (x[j] - mean) * rs;
        float a = nm * qw[c] + qb[c];
        float f = nm * fw[c] + fb[c];
        qnv[j] = a; fnv[j] = f;
        s1 += a; s1q += a * a; s2 += f; s2q += f * f;
        qn_out[(size_t)wid * DIMC + c] = a;
    }
    s1 = wave_sum32(s1); s1q = wave_sum32(s1q);
    s2 = wave_sum32(s2); s2q = wave_sum32(s2q);
    float m1 = s1 * (1.0f / DIMC), r1 = rsqrtf(s1q * (1.0f / DIMC) - m1 * m1 + 1e-6f);
    float m2 = s2 * (1.0f / DIMC), r2 = rsqrtf(s2q * (1.0f / DIMC) - m2 * m2 + 1e-6f);
#pragma unroll
    for (int j = 0; j < 12; ++j) {
        int c = lane + 32 * j;
        aq_out[(size_t)wid * DIMC + c] = (_Float16)((qnv[j] - m1) * r1 * eqw[c] + eqb[c]);
        af_out[(size_t)wid * DIMC + c] = (_Float16)((fnv[j] - m2) * r2 * efw[c] + efb[c]);
    }
}

// ---------------------------------------------------------------------------
// f16 WMMA GEMM: C[M,N] = A[Mpad,K] @ W(col-major, padded [N+,K]) + bias (+resid)
// Block = 8 waves arranged 2(m) x 4(n); each wave computes a 64x16 tile
// (4 WMMAs per K-step sharing one B fragment). All fragment loads are
// unconditional: A has MPAD rows, Wt has padded columns. Stores are guarded.
// Fragment layouts follow CDNA5 ISA 7.12.2.
// ---------------------------------------------------------------------------
__global__ __launch_bounds__(256) void gemm_f16_kernel(
    const _Float16* __restrict__ A, const _Float16* __restrict__ Wt,
    const float* __restrict__ bias, const float* __restrict__ resid,
    float* __restrict__ C, int Mstore, int N, int K) {
    int wave = threadIdx.x >> 5;
    int lane = threadIdx.x & 31;
    int m0 = (blockIdx.x * 2 + (wave >> 2)) * 64;
    int n0 = (blockIdx.y * 4 + (wave & 3)) * 16;
    if (n0 >= N) return;                 // wave-uniform
    int aoff = (lane < 16) ? 0 : 8;      // A: two 8-half runs at +0/+16
    int boff = (lane < 16) ? 0 : 8;      // B: two 8-half runs at +0/+16
    const _Float16* a0 = A + (size_t)(m0 + (lane & 15)) * K + aoff;
    const _Float16* a1 = a0 + (size_t)16 * K;
    const _Float16* a2 = a0 + (size_t)32 * K;
    const _Float16* a3 = a0 + (size_t)48 * K;
    const _Float16* bc = Wt + (size_t)(n0 + (lane & 15)) * K + boff;

    v8f acc0 = {}, acc1 = {}, acc2 = {}, acc3 = {};
    for (int kk = 0; kk < K; kk += 32) {
        union { float4 f4[2]; v16h v; } b, x0, x1, x2, x3;
        b.f4[0]  = *(const float4*)(bc + kk);
        b.f4[1]  = *(const float4*)(bc + kk + 16);
        x0.f4[0] = *(const float4*)(a0 + kk);
        x0.f4[1] = *(const float4*)(a0 + kk + 16);
        x1.f4[0] = *(const float4*)(a1 + kk);
        x1.f4[1] = *(const float4*)(a1 + kk + 16);
        x2.f4[0] = *(const float4*)(a2 + kk);
        x2.f4[1] = *(const float4*)(a2 + kk + 16);
        x3.f4[0] = *(const float4*)(a3 + kk);
        x3.f4[1] = *(const float4*)(a3 + kk + 16);
        if (kk + 32 < K) __builtin_prefetch(a0 + kk + 32, 0, 3);
        acc0 = __builtin_amdgcn_wmma_f32_16x16x32_f16(
            false, x0.v, false, b.v, (short)0, acc0, false, false);
        acc1 = __builtin_amdgcn_wmma_f32_16x16x32_f16(
            false, x1.v, false, b.v, (short)0, acc1, false, false);
        acc2 = __builtin_amdgcn_wmma_f32_16x16x32_f16(
            false, x2.v, false, b.v, (short)0, acc2, false, false);
        acc3 = __builtin_amdgcn_wmma_f32_16x16x32_f16(
            false, x3.v, false, b.v, (short)0, acc3, false, false);
    }
    // C layout: VGPR r -> row m0 + r (+8 for lanes>=16), col n0 + (lane&15)
    int crow0 = m0 + ((lane < 16) ? 0 : 8);
    int ccol = n0 + (lane & 15);
    if (ccol < N) {
        float bv = bias ? bias[ccol] : 0.f;
#define STORE_TILE(ACC, MOFF)                                              \
    _Pragma("unroll") for (int r = 0; r < 8; ++r) {                        \
        int crow = crow0 + (MOFF) + r;                                     \
        if (crow < Mstore) {                                               \
            float v = (ACC)[r] + bv;                                       \
            if (resid) v += resid[(size_t)crow * N + ccol];                \
            C[(size_t)crow * N + ccol] = v;                                \
        }                                                                  \
    }
        STORE_TILE(acc0, 0)
        STORE_TILE(acc1, 16)
        STORE_TILE(acc2, 32)
        STORE_TILE(acc3, 48)
#undef STORE_TILE
    }
}

// ---------------------------------------------------------------------------
// Softmax over the 12 (level,point) logits per (row, head)
// ---------------------------------------------------------------------------
__global__ void softmax12_kernel(float* __restrict__ aw) {
    int t = blockIdx.x * blockDim.x + threadIdx.x;
    if (t >= MROWS * HEADS) return;
    float* p = aw + (size_t)t * 12;
    float mx = p[0];
#pragma unroll
    for (int i = 1; i < 12; ++i) mx = fmaxf(mx, p[i]);
    float e[12]; float s = 0.f;
#pragma unroll
    for (int i = 0; i < 12; ++i) { e[i] = __expf(p[i] - mx); s += e[i]; }
    float inv = 1.0f / s;
#pragma unroll
    for (int i = 0; i < 12; ++i) p[i] = e[i] * inv;
}

// ---------------------------------------------------------------------------
// Deformable trilinear sampling + attention reduce. One wave per (row, head);
// each lane owns channels 2*lane, 2*lane+1 of HDIM=64. Output f16 [M, 384].
// ---------------------------------------------------------------------------
__global__ __launch_bounds__(256) void deform_sample_kernel(
    const float* __restrict__ value, const float* __restrict__ off,
    const float* __restrict__ aw, _Float16* __restrict__ dout) {
    int wid = blockIdx.x * (blockDim.x >> 5) + (threadIdx.x >> 5);
    if (wid >= MROWS * HEADS) return;
    int lane = threadIdx.x & 31;
    int row = wid / HEADS, head = wid - row * HEADS;
    int b = row / NTOK, s = row - b * NTOK;

    // reference point from the token's own level (d, x, y)
    int Dq, t;
    if (s < NHIGH)      { Dq = 24; t = s; }
    else if (s < NMIDE) { Dq = 12; t = s - NHIGH; }
    else                { Dq = 6;  t = s - NMIDE; }
    int td = t / (Dq * Dq);
    int rem = t - td * Dq * Dq;
    int ty = rem / Dq, tx = rem - ty * Dq;
    float invq = 1.0f / (float)Dq;
    float ref_d = ((float)td + 0.5f) * invq;
    float ref_x = ((float)tx + 0.5f) * invq;
    float ref_y = ((float)ty + 0.5f) * invq;

    const int   lvl_dim[3]   = {24, 12, 6};
    const int   lvl_start[3] = {0, NHIGH, NMIDE};
    const float* orow = off + (size_t)row * (HEADS * LEVELS * POINTS * 3);
    const float* arow = aw + (size_t)row * (HEADS * LEVELS * POINTS);

    float acc0 = 0.f, acc1 = 0.f;
#pragma unroll
    for (int l = 0; l < LEVELS; ++l) {
        int Dl = lvl_dim[l];
        float fDl = (float)Dl, inv_l = 1.0f / fDl;
        const float* vbase = value + ((size_t)b * NTOK + lvl_start[l]) * DIMC
                             + head * HDIM + lane * 2;
#pragma unroll
        for (int p = 0; p < POINTS; ++p) {
            const float* o = orow + ((head * LEVELS + l) * POINTS + p) * 3;
            float watt = arow[(head * LEVELS + l) * POINTS + p];
            float ld = ref_d + o[0] * inv_l;
            float lx = ref_x + o[1] * inv_l;
            float ly = ref_y + o[2] * inv_l;
            float pd = ld * fDl - 0.5f;
            float px = lx * fDl - 0.5f;
            float py = ly * fDl - 0.5f;
            float fd0 = floorf(pd), fx0 = floorf(px), fy0 = floorf(py);
            float fd = pd - fd0, fx = px - fx0, fy = py - fy0;
            int d0 = (int)fd0, x0 = (int)fx0, y0 = (int)fy0;
#pragma unroll
            for (int cd = 0; cd < 2; ++cd)
#pragma unroll
                for (int cy = 0; cy < 2; ++cy)
#pragma unroll
                    for (int cx = 0; cx < 2; ++cx) {
                        int di = d0 + cd, yi = y0 + cy, xi = x0 + cx;
                        bool valid = (di >= 0) & (di < Dl) & (yi >= 0) & (yi < Dl) &
                                     (xi >= 0) & (xi < Dl);
                        float wgt = (cd ? fd : 1.f - fd) * (cy ? fy : 1.f - fy) *
                                    (cx ? fx : 1.f - fx);
                        if (valid && wgt != 0.f) {
                            int idx = (di * Dl + yi) * Dl + xi;
                            const float2 g = *(const float2*)(vbase + (size_t)idx * DIMC);
                            float wv = wgt * watt;
                            acc0 += wv * g.x;
                            acc1 += wv * g.y;
                        }
                    }
        }
    }
    _Float16* dp = dout + (size_t)row * DIMC + head * HDIM + lane * 2;
    dp[0] = (_Float16)acc0;
    dp[1] = (_Float16)acc1;
}

// ---------------------------------------------------------------------------
// Single layernorm -> f16 (FFN input). One wave per row.
// ---------------------------------------------------------------------------
__global__ __launch_bounds__(256) void ln_f16_kernel(
    const float* __restrict__ x, const float* __restrict__ w,
    const float* __restrict__ bb, _Float16* __restrict__ y) {
    int wid = blockIdx.x * (blockDim.x >> 5) + (threadIdx.x >> 5);
    if (wid >= MROWS) return;
    int lane = threadIdx.x & 31;
    const float* xr = x + (size_t)wid * DIMC;
    float v[12]; float sum = 0.f, sq = 0.f;
#pragma unroll
    for (int j = 0; j < 12; ++j) {
        float t = xr[lane + 32 * j];
        v[j] = t; sum += t; sq += t * t;
    }
    sum = wave_sum32(sum); sq = wave_sum32(sq);
    float m = sum * (1.0f / DIMC);
    float rs = rsqrtf(sq * (1.0f / DIMC) - m * m + 1e-6f);
#pragma unroll
    for (int j = 0; j < 12; ++j) {
        int c = lane + 32 * j;
        y[(size_t)wid * DIMC + c] = (_Float16)((v[j] - m) * rs * w[c] + bb[c]);
    }
}

// ---------------------------------------------------------------------------
// Depthwise 3x3x3 conv (SAME, zero pad) on the three spatial segments + GELU.
// One thread per (row, channel). Output f16 [M, 96].
// ---------------------------------------------------------------------------
__global__ void dwconv_gelu_kernel(const float* __restrict__ x,
                                   const float* __restrict__ dw,
                                   const float* __restrict__ db,
                                   _Float16* __restrict__ y) {
    int t = blockIdx.x * blockDim.x + threadIdx.x;
    if (t >= MROWS * HIDC) return;
    int row = t / HIDC, ch = t - row * HIDC;
    int b = row / NTOK, s = row - b * NTOK;
    int Dv, base;
    if (s < NHIGH)      { Dv = 24; base = 0; }
    else if (s < NMIDE) { Dv = 12; base = NHIGH; }
    else                { Dv = 6;  base = NMIDE; }
    int ti = s - base;
    int d = ti / (Dv * Dv);
    int rem = ti - d * Dv * Dv;
    int h = rem / Dv, w = rem - h * Dv;
    const float* xseg = x + ((size_t)b * NTOK + base) * HIDC + ch;
    float acc = db[ch];
#pragma unroll
    for (int kd = 0; kd < 3; ++kd) {
        int dd = d + kd - 1;
        if (dd < 0 || dd >= Dv) continue;
#pragma unroll
        for (int kh = 0; kh < 3; ++kh) {
            int hh = h + kh - 1;
            if (hh < 0 || hh >= Dv) continue;
#pragma unroll
            for (int kw = 0; kw < 3; ++kw) {
                int ww = w + kw - 1;
                if (ww < 0 || ww >= Dv) continue;
                int nb = (dd * Dv + hh) * Dv + ww;
                acc += xseg[(size_t)nb * HIDC] *
                       dw[((kd * 3 + kh) * 3 + kw) * HIDC + ch];
            }
        }
    }
    float g = 0.5f * acc * (1.0f + erff(acc * 0.70710678118654752f));
    y[(size_t)row * HIDC + ch] = (_Float16)g;
}

// ---------------------------------------------------------------------------
extern "C" void kernel_launch(void* const* d_in, const int* in_sizes, int n_in,
                              void* d_out, int out_size, void* d_ws, size_t ws_size,
                              hipStream_t stream) {
    (void)in_sizes; (void)n_in; (void)out_size; (void)ws_size;
    const float* query  = (const float*)d_in[0];
    const float* feat   = (const float*)d_in[2];
    const float* qnorm_w = (const float*)d_in[8],  *qnorm_b = (const float*)d_in[9];
    const float* fnorm_w = (const float*)d_in[10], *fnorm_b = (const float*)d_in[11];
    const float* eq_w = (const float*)d_in[12], *eq_b = (const float*)d_in[13];
    const float* ef_w = (const float*)d_in[14], *ef_b = (const float*)d_in[15];
    const float* ffn_w = (const float*)d_in[16], *ffn_b = (const float*)d_in[17];
    const float* off_w = (const float*)d_in[18], *off_b = (const float*)d_in[19];
    const float* aw_w  = (const float*)d_in[20], *aw_b  = (const float*)d_in[21];
    const float* val_w = (const float*)d_in[22], *val_b = (const float*)d_in[23];
    const float* out_w = (const float*)d_in[24], *out_b = (const float*)d_in[25];
    const float* fc1_w = (const float*)d_in[26], *fc1_b = (const float*)d_in[27];
    const float* dw_w  = (const float*)d_in[28], *dw_b  = (const float*)d_in[29];
    const float* fc2_w = (const float*)d_in[30], *fc2_b = (const float*)d_in[31];
    float* out = (float*)d_out;

    // ----- workspace carve-up (256B aligned); activations padded to MPAD rows
    char* p = (char*)d_ws;
    auto alloc = [&](size_t bytes) -> void* {
        void* r = (void*)p;
        p += (bytes + 255) & ~(size_t)255;
        return r;
    };
    const size_t MP = MPAD;
    float*    qn     = (float*)   alloc(MP * DIMC * 4);
    _Float16* aq     = (_Float16*)alloc(MP * DIMC * 2);
    _Float16* af     = (_Float16*)alloc(MP * DIMC * 2);
    float*    value  = (float*)   alloc(MP * DIMC * 4);
    float*    offb   = (float*)   alloc(MP * 216 * 4);
    float*    awb    = (float*)   alloc(MP * 72 * 4);
    _Float16* dsamp  = (_Float16*)alloc(MP * DIMC * 2);
    float*    out1   = (float*)   alloc(MP * DIMC * 4);
    _Float16* ffn_in = (_Float16*)alloc(MP * DIMC * 2);
    float*    fc1o   = (float*)   alloc(MP * HIDC * 4);
    _Float16* act    = (_Float16*)alloc(MP * HIDC * 2);
    // transposed weights, padded by 64 extra columns so B-tile loads never guard
    _Float16* w_val  = (_Float16*)alloc((size_t)(DIMC + 64) * DIMC * 2);
    _Float16* w_off  = (_Float16*)alloc((size_t)(216 + 64) * DIMC * 2);
    _Float16* w_aw   = (_Float16*)alloc((size_t)(72 + 64) * DIMC * 2);
    _Float16* w_out  = (_Float16*)alloc((size_t)(DIMC + 64) * DIMC * 2);
    _Float16* w_fc1  = (_Float16*)alloc((size_t)(HIDC + 64) * DIMC * 2);
    _Float16* w_fc2  = (_Float16*)alloc((size_t)(DIMC + 64) * HIDC * 2);

    // ----- 0) weight conversion / transpose -----
    auto cvt = [&](const float* w, _Float16* wt, int K, int N) {
        int total = K * N;
        convert_wt_kernel<<<(total + 255) / 256, 256, 0, stream>>>(w, wt, K, N);
    };
    cvt(val_w, w_val, DIMC, DIMC);
    cvt(off_w, w_off, DIMC, 216);
    cvt(aw_w,  w_aw,  DIMC, 72);
    cvt(out_w, w_out, DIMC, DIMC);
    cvt(fc1_w, w_fc1, DIMC, HIDC);
    cvt(fc2_w, w_fc2, HIDC, DIMC);

    // ----- 1) fused prep: q build + 4 layernorms -----
    prep_kernel<<<(MROWS + 7) / 8, 256, 0, stream>>>(
        query, feat, qnorm_w, qnorm_b, fnorm_w, fnorm_b,
        eq_w, eq_b, ef_w, ef_b, qn, aq, af);

    // ----- 2) WMMA GEMMs: value / offsets / attention logits -----
    auto gemm = [&](const _Float16* A, const _Float16* Wt, const float* bias,
                    const float* resid, float* C, int N, int K) {
        dim3 grid(MPAD / 128, (N + 63) / 64);
        gemm_f16_kernel<<<grid, 256, 0, stream>>>(A, Wt, bias, resid, C,
                                                  MROWS, N, K);
    };
    gemm(af, w_val, val_b, nullptr, value, DIMC, DIMC);
    gemm(aq, w_off, off_b, nullptr, offb, 216, DIMC);
    gemm(aq, w_aw,  aw_b,  nullptr, awb, 72, DIMC);

    // ----- 3) softmax + deformable sampling -----
    softmax12_kernel<<<(MROWS * HEADS + 255) / 256, 256, 0, stream>>>(awb);
    deform_sample_kernel<<<(MROWS * HEADS + 7) / 8, 256, 0, stream>>>(
        value, offb, awb, dsamp);

    // ----- 4) output projection (+qn residual) -----
    gemm(dsamp, w_out, out_b, qn, out1, DIMC, DIMC);

    // ----- 5) conv-FFN -----
    ln_f16_kernel<<<(MROWS + 7) / 8, 256, 0, stream>>>(out1, ffn_w, ffn_b, ffn_in);
    gemm(ffn_in, w_fc1, fc1_b, nullptr, fc1o, HIDC, DIMC);
    dwconv_gelu_kernel<<<(MROWS * HIDC + 255) / 256, 256, 0, stream>>>(
        fc1o, dw_w, dw_b, act);
    gemm(act, w_fc2, fc2_b, out1, out, DIMC, HIDC);
}
